// img_as_hyper_tabMLP_AGE1to3_53369263620494
// MI455X (gfx1250) — matrix operand order repaired and checked by reading, not compiled, not run on hardware
//
#include <hip/hip_runtime.h>
#include <hip/hip_bf16.h>

typedef _Float16 f16;
typedef __attribute__((ext_vector_type(16))) _Float16 v16h;
typedef __attribute__((ext_vector_type(8)))  _Float16 v8h;
typedef __attribute__((ext_vector_type(8)))  float    v8f;
typedef int v4i __attribute__((vector_size(16)));

#define EPSV 1e-5f

#ifndef __has_builtin
#define __has_builtin(x) 0
#endif
#if __has_builtin(__builtin_amdgcn_global_load_async_to_lds_b128)
#define HAVE_ASYNC_LDS 1
#else
#define HAVE_ASYNC_LDS 0
#endif
#if __has_builtin(__builtin_amdgcn_s_wait_asynccnt)
#define HAVE_WAIT_ASYNC 1
#else
#define HAVE_WAIT_ASYNC 0
#endif

typedef __attribute__((address_space(1))) v4i* as1_v4i;
typedef __attribute__((address_space(3))) v4i* as3_v4i;

// 16-byte global->LDS copy. Async (GLOBAL_LOAD_ASYNC_TO_LDS_B128, ASYNCcnt)
// when the builtin exists, else a synchronous ds store fallback.
__device__ __forceinline__ void copy16_g2l(const f16* g, f16* l) {
#if HAVE_ASYNC_LDS
  __builtin_amdgcn_global_load_async_to_lds_b128(
      (as1_v4i)(unsigned long long)g,
      (as3_v4i)(unsigned)(unsigned long long)l, 0, 0);
#else
  *(v8h*)l = *(const v8h*)g;
#endif
}

__device__ __forceinline__ void wait_async_done() {
#if HAVE_ASYNC_LDS
#if HAVE_WAIT_ASYNC
  __builtin_amdgcn_s_wait_asynccnt(0);
#else
  asm volatile("s_wait_asynccnt 0" ::: "memory");
#endif
#endif
}

// ---------------------------------------------------------------------------
// Pack fp32 weights [Cout][K] (natural OIDHW flattening: k = cin*27 + tap)
// into fp16 [Cout][Kp], zero-padded so the WMMA K-loop is exact & aligned.
// ---------------------------------------------------------------------------
__global__ void k_pack_w(const float* __restrict__ w, f16* __restrict__ o,
                         int Cout, int K, int Kp) {
  int i = blockIdx.x * 256 + threadIdx.x;
  int total = Cout * Kp;
  if (i >= total) return;
  int co = i / Kp;
  int k  = i - co * Kp;
  o[i] = (k < K) ? (f16)w[co * K + k] : (f16)0.f;
}

// ---------------------------------------------------------------------------
// y = fp16( relu(x * g/sqrt(1+eps) + b) ); if g==null, plain fp32->fp16 cvt.
// ---------------------------------------------------------------------------
__global__ void k_bnrelu_cvt(const float* __restrict__ x, const float* __restrict__ g,
                             const float* __restrict__ bb, f16* __restrict__ y,
                             int C, int S, int n) {
  int i = blockIdx.x * 256 + threadIdx.x;
  if (i >= n) return;
  float v = x[i];
  if (g) {
    int c = (i / S) % C;
    v = v * (g[c] * rsqrtf(1.f + EPSV)) + bb[c];
    v = v > 0.f ? v : 0.f;
  }
  y[i] = (f16)v;
}

// ---------------------------------------------------------------------------
// Implicit-GEMM 3D conv via V_WMMA_F32_16X16X32_F16 (wave32).
//   GEMM: M = Cout, N = 8*Dout^3, K = Cin*TAPS (Kp padded to mult of 32)
//   block = 128 threads = 4 waves; block tile = (16*MBLK) x 64.
//   Wave w owns N-subtile (blockIdx.x*4+w)*16; all waves share M rows.
// Weight tile (MBLK*16 rows x 32 k, fp16) is staged per k-chunk into LDS with
// async global->LDS b128 copies (ASYNCcnt + barrier), then each wave builds
// MBLK A fragments from LDS (ds_load_b128 pairs, 16-bit A 16x32 layout:
//   lane(m=lane&15,h=lane>>4): e0..7 = K[h*8..+7], e8..15 = K[16+h*8..+7])
// and one B fragment by im2col gather (B 32x16 layout: lane(n,h) holds
//   e0..7 = K[h*8..+7], e8..15 = K[16+h*8..+7] of column n), amortized over MBLK WMMAs.
// C/D: VGPR r <-> (M = r + 8*h, N = lane&15).
// mode 0: out32 = acc (+resid).  mode 1: out16 = relu(acc*s+b)  (fused BN).
// ---------------------------------------------------------------------------
template <int TAPS, int MBLK>
__global__ __launch_bounds__(128) void k_conv_wmma(
    const f16* __restrict__ act, const f16* __restrict__ wgt,
    float* __restrict__ out32, f16* __restrict__ out16,
    const float* __restrict__ resid,
    const float* __restrict__ bng, const float* __restrict__ bnb,
    int Cin, int Cout, int Din, int Dout, int stride, int Kp, int mode) {
  __shared__ __align__(16) f16 Atile[MBLK * 16 * 32];

  const int tid  = threadIdx.x;
  const int lane = tid & 31;
  const int wave = tid >> 5;
  const int hh   = lane >> 4;   // half-wave index
  const int nm   = lane & 15;   // B col (n) / A row-in-tile (m)

  const int Dout2 = Dout * Dout;
  const int Dout3 = Dout2 * Dout;
  const int col   = (blockIdx.x * 4 + wave) * 16 + nm;   // output column
  const int bi    = col / Dout3;
  const int sp    = col - bi * Dout3;
  const int z     = sp / Dout2;
  const int y     = (sp - z * Dout2) / Dout;
  const int x     = sp - z * Dout2 - y * Dout;
  const int pad   = (TAPS == 27) ? 1 : 0;
  const int zi0 = z * stride - pad, yi0 = y * stride - pad, xi0 = x * stride - pad;
  const int Din2 = Din * Din;
  const f16* actb = act + (size_t)bi * Cin * Din * Din2;
  const int K = Cin * TAPS;
  const int cout0 = blockIdx.y * (16 * MBLK);

  v8f acc[MBLK] = {};

  for (int kc = 0; kc < Kp; kc += 32) {
    if (kc) __syncthreads();          // previous chunk fully consumed
    // ---- stage A tile: MBLK*32 16-byte units; unit u -> (row=u>>1, hf=u&1)
#pragma unroll
    for (int u0 = 0; u0 < MBLK * 32; u0 += 128) {
      int u = u0 + tid;
      if (u < MBLK * 32) {
        int row = u >> 1, hf = u & 1;
        copy16_g2l(wgt + (size_t)(cout0 + row) * Kp + kc + hf * 16,
                   &Atile[u * 16]);
      }
    }
    wait_async_done();
    __syncthreads();                  // tile visible to all waves

    // ---- B fragment: im2col gather of 16 fp16 values ----
    v16h bf;
#pragma unroll
    for (int e = 0; e < 16; e++) {
      int k = kc + hh * 8 + e + ((e >= 8) ? 8 : 0);
      f16 v = (f16)0.f;
      if (k < K) {
        int ci = k / TAPS;
        int r  = k - ci * TAPS;
        int kd = r / 9; int rr = r - kd * 9; int kh = rr / 3; int kw = rr - kh * 3;
        int zi = zi0 + kd, yi = yi0 + kh, xi = xi0 + kw;
        if (((unsigned)zi < (unsigned)Din) && ((unsigned)yi < (unsigned)Din) &&
            ((unsigned)xi < (unsigned)Din))
          v = actb[((size_t)ci * Din + zi) * Din2 + yi * Din + xi];
      }
      bf[e] = v;
    }

    // ---- MBLK A fragments from LDS, MBLK WMMAs sharing bf ----
#pragma unroll
    for (int mi = 0; mi < MBLK; mi++) {
      const f16* ar = &Atile[(mi * 16 + nm) * 32 + hh * 8];
      v8h alo = *(const v8h*)ar;         // ds_load_b128
      v8h ahi = *(const v8h*)(ar + 16);  // ds_load_b128
      v16h a;
#pragma unroll
      for (int i = 0; i < 8; i++) { a[i] = alo[i]; a[i + 8] = ahi[i]; }
      acc[mi] = __builtin_amdgcn_wmma_f32_16x16x32_f16(false, a, false, bf,
                                                       (short)0, acc[mi],
                                                       false, false);
    }
  }

  // ---- epilogue ----
#pragma unroll
  for (int mi = 0; mi < MBLK; mi++) {
#pragma unroll
    for (int r = 0; r < 8; r++) {
      int co = cout0 + mi * 16 + hh * 8 + r;
      size_t off = ((size_t)bi * Cout + co) * Dout3 + sp;
      float v = acc[mi][r];
      if (resid) v += resid[off];
      if (mode == 1) {
        v = v * (bng[co] * rsqrtf(1.f + EPSV)) + bnb[co];
        v = v > 0.f ? v : 0.f;
        out16[off] = (f16)v;
      } else {
        out32[off] = v;
      }
    }
  }
}

// ---------------------------------------------------------------------------
// feat[b][c] = mean_s relu( x[b][c][s]*g[c]/sqrt(1+eps) + b[c] ),  S = 4^3
// ---------------------------------------------------------------------------
__global__ void k_meanbn(const float* __restrict__ x, const float* __restrict__ g,
                         const float* __restrict__ bb, float* __restrict__ feat,
                         int C, int S) {
  int i = blockIdx.x * 256 + threadIdx.x;   // i = b*C + c
  if (i >= 8 * C) return;
  int c = i % C;
  float s = g[c] * rsqrtf(1.f + EPSV);
  float bias = bb[c];
  const float* ptr = x + (size_t)i * S;
  float a = 0.f;
  for (int j = 0; j < S; j++) { float v = ptr[j] * s + bias; a += (v > 0.f) ? v : 0.f; }
  feat[i] = a / (float)S;
}

// ---------------------------------------------------------------------------
// Head: fc1(512->64)+relu, fc2(64->32), gen(32->6), hyper-linear w/ tabular.
// out[b][o] = fcp[b][o]*tab[b] + fcp[b][3+o]   (N_TAB=1, N_OUT=3)
// ---------------------------------------------------------------------------
__global__ void k_head(const float* __restrict__ feat,
                       const float* __restrict__ fc1w, const float* __restrict__ fc1b,
                       const float* __restrict__ fc2w, const float* __restrict__ fc2b,
                       const float* __restrict__ genw, const float* __restrict__ genb,
                       const float* __restrict__ tab, float* __restrict__ out) {
  __shared__ float h1[8][64];
  __shared__ float h2[8][32];
  __shared__ float fp[8][6];
  int t = threadIdx.x;  // 64 threads
  for (int b = 0; b < 8; b++) {
    float a = fc1b[t];
    for (int c = 0; c < 512; c++) a += feat[b * 512 + c] * fc1w[t * 512 + c];
    h1[b][t] = a > 0.f ? a : 0.f;
  }
  __syncthreads();
  if (t < 32) for (int b = 0; b < 8; b++) {
    float a = fc2b[t];
    for (int c = 0; c < 64; c++) a += h1[b][c] * fc2w[t * 64 + c];
    h2[b][t] = a;
  }
  __syncthreads();
  if (t < 6) for (int b = 0; b < 8; b++) {
    float a = genb[t];
    for (int c = 0; c < 32; c++) a += h2[b][c] * genw[t * 32 + c];
    fp[b][t] = a;
  }
  __syncthreads();
  if (t < 24) { int b = t / 3, o = t - b * 3; out[t] = fp[b][o] * tab[b] + fp[b][3 + o]; }
}

// ---------------------------------------------------------------------------
static void conv_launch(int taps, int mblk, const f16* act, const f16* wgt,
                        float* out32, f16* out16, const float* resid,
                        const float* g, const float* bb,
                        int Cin, int Cout, int Din, int Dout, int stride,
                        int Kp, int mode, hipStream_t s) {
  long Ncols = 8L * Dout * Dout * Dout;
  dim3 grid((unsigned)(Ncols / 64), (unsigned)(Cout / (16 * mblk)), 1);
  if (taps == 27 && mblk == 4)
    k_conv_wmma<27, 4><<<grid, 128, 0, s>>>(act, wgt, out32, out16, resid, g, bb,
                                            Cin, Cout, Din, Dout, stride, Kp, mode);
  else if (taps == 27)
    k_conv_wmma<27, 2><<<grid, 128, 0, s>>>(act, wgt, out32, out16, resid, g, bb,
                                            Cin, Cout, Din, Dout, stride, Kp, mode);
  else
    k_conv_wmma<1, 4><<<grid, 128, 0, s>>>(act, wgt, out32, out16, resid, g, bb,
                                           Cin, Cout, Din, Dout, stride, Kp, mode);
}

extern "C" void kernel_launch(void* const* d_in, const int* in_sizes, int n_in,
                              void* d_out, int out_size, void* d_ws, size_t ws_size,
                              hipStream_t stream) {
  (void)in_sizes; (void)n_in; (void)out_size; (void)ws_size;
  // ---- unpack inputs (setup_inputs dict order, params flattened in order) ----
  const float* image = (const float*)d_in[0];
  const float* tab   = (const float*)d_in[1];
  const float* stemw = (const float*)d_in[2];
  const float *bn1g[4], *bn1b[4], *c1w[4], *bn2g[4], *bn2b[4], *c2w[4], *scw[4];
  int idx = 3;
  for (int i = 0; i < 4; i++) {
    bn1g[i] = (const float*)d_in[idx++]; bn1b[i] = (const float*)d_in[idx++];
    c1w[i]  = (const float*)d_in[idx++];
    bn2g[i] = (const float*)d_in[idx++]; bn2b[i] = (const float*)d_in[idx++];
    c2w[i]  = (const float*)d_in[idx++]; scw[i]  = (const float*)d_in[idx++];
  }
  const float* bnfg = (const float*)d_in[idx++];
  const float* bnfb = (const float*)d_in[idx++];
  const float* fc1w = (const float*)d_in[idx++];
  const float* fc1b = (const float*)d_in[idx++];
  const float* fc2w = (const float*)d_in[idx++];
  const float* fc2b = (const float*)d_in[idx++];
  const float* genw = (const float*)d_in[idx++];
  const float* genb = (const float*)d_in[idx++];

  // ---- workspace bump allocator (~335 MB peak) ----
  char* p = (char*)d_ws;
  auto carve = [&](size_t bytes) -> void* {
    void* r = (void*)p; p += (bytes + 255) & ~(size_t)255; return r;
  };
  const int CIN[4]  = {32, 64, 128, 256};
  const int COUT[4] = {64, 128, 256, 512};
  const int DIN[4]  = {64, 32, 16, 8};
  const int DOUTA[4]= {32, 16, 8, 4};
  f16* stemW16 = (f16*)carve((size_t)32 * 32 * sizeof(f16));
  f16 *C1W[4], *C2W[4], *SCW[4];
  for (int i = 0; i < 4; i++) {
    C1W[i] = (f16*)carve((size_t)COUT[i] * CIN[i] * 27 * sizeof(f16));
    C2W[i] = (f16*)carve((size_t)COUT[i] * COUT[i] * 27 * sizeof(f16));
    SCW[i] = (f16*)carve((size_t)COUT[i] * CIN[i] * sizeof(f16));
  }
  f16*   img16 = (f16*)carve((size_t)8 * 64 * 64 * 64 * sizeof(f16));
  f16*   H1 = (f16*)carve((size_t)8 * 32 * 64 * 64 * 64 * sizeof(f16));     // 134 MB
  float* SC = (float*)carve((size_t)8 * 64 * 32 * 32 * 32 * sizeof(float)); // 67 MB
  f16*   H2 = (f16*)carve((size_t)8 * 64 * 32 * 32 * 32 * sizeof(f16));     // 34 MB
  float* X  = (float*)carve((size_t)8 * 64 * 32 * 32 * 32 * sizeof(float)); // 67 MB
  float* feat = (float*)carve((size_t)8 * 512 * sizeof(float));

  // ---- pack all weights to fp16 ----
  {
    int total = 32 * 32;
    k_pack_w<<<(total + 255) / 256, 256, 0, stream>>>(stemw, stemW16, 32, 27, 32);
  }
  for (int i = 0; i < 4; i++) {
    int t1 = COUT[i] * CIN[i] * 27;
    k_pack_w<<<(t1 + 255) / 256, 256, 0, stream>>>(c1w[i], C1W[i], COUT[i], CIN[i] * 27, CIN[i] * 27);
    int t2 = COUT[i] * COUT[i] * 27;
    k_pack_w<<<(t2 + 255) / 256, 256, 0, stream>>>(c2w[i], C2W[i], COUT[i], COUT[i] * 27, COUT[i] * 27);
    int t3 = COUT[i] * CIN[i];
    k_pack_w<<<(t3 + 255) / 256, 256, 0, stream>>>(scw[i], SCW[i], COUT[i], CIN[i], CIN[i]);
  }
  // ---- image fp32 -> fp16 ----
  {
    int n = 8 * 64 * 64 * 64;
    k_bnrelu_cvt<<<(n + 255) / 256, 256, 0, stream>>>(image, nullptr, nullptr, img16, 1, 1, n);
  }
  // ---- stem conv, fused with block0 bn1+relu -> H1 (fp16); Cout=32 -> MBLK=2
  conv_launch(27, 2, img16, stemW16, nullptr, H1, nullptr, bn1g[0], bn1b[0],
              1, 32, 64, 64, 1, 32, 1, stream);
  // ---- residual blocks ----
  for (int i = 0; i < 4; i++) {
    int cin = CIN[i], cout = COUT[i], din = DIN[i], dout = DOUTA[i];
    if (i > 0) {  // h1 = relu(bn1(x)) -> fp16
      int n = 8 * cin * din * din * din;
      k_bnrelu_cvt<<<(n + 255) / 256, 256, 0, stream>>>(X, bn1g[i], bn1b[i], H1,
                                                        cin, din * din * din, n);
    }
    // shortcut: 1x1x1 stride-2 conv -> SC (fp32)
    conv_launch(1, 4, H1, SCW[i], SC, nullptr, nullptr, nullptr, nullptr,
                cin, cout, din, dout, 2, cin, 0, stream);
    // conv1: 3x3x3 stride-2, fused bn2+relu -> H2 (fp16)
    conv_launch(27, 4, H1, C1W[i], nullptr, H2, nullptr, bn2g[i], bn2b[i],
                cin, cout, din, dout, 2, cin * 27, 1, stream);
    // conv2: 3x3x3 stride-1, + shortcut residual -> X (fp32)
    conv_launch(27, 4, H2, C2W[i], X, nullptr, SC, nullptr, nullptr,
                cout, cout, dout, dout, 1, cout * 27, 0, stream);
  }
  // ---- final bn+relu+mean ----
  {
    int n = 8 * 512;
    k_meanbn<<<(n + 255) / 256, 256, 0, stream>>>(X, bnfg, bnfb, feat, 512, 64);
  }
  // ---- hypernetwork head ----
  k_head<<<1, 64, 0, stream>>>(feat, fc1w, fc1b, fc2w, fc2b, genw, genb, tab,
                               (float*)d_out);
}
